// AlphaNet_17231408792021
// MI455X (gfx1250) — compile-verified, matrix operand order checked by reading, not run on hardware
//
#include <hip/hip_runtime.h>
#include <hip/hip_bf16.h>
#include <stdint.h>

// ---------------- types & helpers ----------------
typedef __attribute__((ext_vector_type(16))) __bf16 v16bf;
typedef __attribute__((ext_vector_type(8)))  float  v8f;

#define INV_SQRT_3f 0.5773502691896258f
#define INV_SQRT_Hf 0.08838834764831845f   /* 1/sqrt(128) */
#define KSCALEf     0.3535533905932738f    /* 1/sqrt(8)   */

__device__ __forceinline__ __bf16 f2bf(float f) {
    unsigned u = __float_as_uint(f);
    unsigned short s = (unsigned short)((u + 0x7FFFu + ((u >> 16) & 1u)) >> 16);
    return __builtin_bit_cast(__bf16, s);
}
__device__ __forceinline__ float silu(float x) { return x / (1.f + __expf(-x)); }

struct U4x2 { uint4 a, b; };
// Load a 16-element bf16 fragment: two contiguous 16B runs ({k..k+7},{k+16..k+23})
__device__ __forceinline__ v16bf load_frag(const __bf16* p) {
    U4x2 u;
    u.a = *(const uint4*)(p);
    u.b = *(const uint4*)(p + 16);
    return __builtin_bit_cast(v16bf, u);
}
__device__ __forceinline__ v8f zero8() {
    v8f z;
#pragma unroll
    for (int i = 0; i < 8; i++) z[i] = 0.f;
    return z;
}
__device__ __forceinline__ v8f wmma_bf16(v16bf a, v16bf b, v8f c) {
    return __builtin_amdgcn_wmma_f32_16x16x32_bf16(false, a, false, b, (short)0, c, false, false);
}

// ---------------- param prep ----------------
// W (K x Nc, f32 row-major) -> WT (Nc x K, bf16 row-major)
__global__ void k_cvt_t(const float* __restrict__ W, __bf16* __restrict__ WT, int K, int Nc) {
    int i = blockIdx.x * 256 + threadIdx.x;
    if (i >= K * Nc) return;
    int k = i / Nc, n = i - k * Nc;
    WT[(size_t)n * K + k] = f2bf(W[i]);
}

// u = diachi1 @ dia_w ; v = colsum(dia_w)+dia_b ; U = u@fc_w ; V = v@fc_w  (32 threads)
__global__ void k_uv(const float* __restrict__ diachi1, const float* __restrict__ dia_w,
                     const float* __restrict__ dia_b, const float* __restrict__ fc_w,
                     float* __restrict__ UV) {
    __shared__ float u[32], v[32];
    int t = threadIdx.x;
    float su = 0.f, sv = 0.f;
    for (int l = 0; l < 32; l++) { float w = dia_w[l * 32 + t]; su += diachi1[l] * w; sv += w; }
    u[t] = su; v[t] = sv + dia_b[t];
    __syncthreads();
    float Uo = 0.f, Vo = 0.f;
    for (int l = 0; l < 32; l++) { float f = fc_w[l * 32 + t]; Uo += u[l] * f; Vo += v[l] * f; }
    UV[t] = Uo; UV[32 + t] = Vo;
}

// ---------------- node MLP: xh = silu(LN(x)@w1+b1)@w2+b2  (16 rows / block) ----------------
__global__ __launch_bounds__(128) void k_node(
    const float* __restrict__ x, const float* __restrict__ ln_g, const float* __restrict__ ln_b,
    const __bf16* __restrict__ w1t, const float* __restrict__ b1,
    const __bf16* __restrict__ w2t, const float* __restrict__ b2,
    float* __restrict__ xh, int N) {
    __shared__ __align__(16) float  xs[16 * 128];
    __shared__ __align__(16) __bf16 xa[16 * 128];
    __shared__ float mu[16], rs[16];
    int tid = threadIdx.x, r0 = blockIdx.x * 16;
    for (int i = tid; i < 2048; i += 128) {
        int r = i >> 7, c = i & 127, g = r0 + r;
        xs[i] = (g < N) ? x[(size_t)g * 128 + c] : 0.f;
    }
    __syncthreads();
    if (tid < 16) {
        float m = 0.f;
        for (int c = 0; c < 128; c++) m += xs[tid * 128 + c];
        m *= (1.f / 128.f);
        float v = 0.f;
        for (int c = 0; c < 128; c++) { float d = xs[tid * 128 + c] - m; v += d * d; }
        mu[tid] = m; rs[tid] = rsqrtf(v * (1.f / 128.f) + 1e-5f);
    }
    __syncthreads();
    for (int i = tid; i < 2048; i += 128) {
        int r = i >> 7, c = i & 127;
        xa[i] = f2bf((xs[i] - mu[r]) * rs[r] * ln_g[c] + ln_b[c]);
    }
    __syncthreads();
    int lane = tid & 31, w = tid >> 5, lm = lane & 15, kh = (lane >> 4) * 8, mh = (lane >> 4) * 8;
    // GEMM1: 16x128 @ 128x128, 8 col tiles -> 2 per wave
    v8f a1[2]; a1[0] = zero8(); a1[1] = zero8();
    for (int kt = 0; kt < 4; kt++) {
        v16bf af = load_frag(xa + lm * 128 + kt * 32 + kh);
#pragma unroll
        for (int i = 0; i < 2; i++) {
            int n = (w * 2 + i) * 16 + lm;
            v16bf bf_ = load_frag(w1t + (size_t)n * 128 + kt * 32 + kh);
            a1[i] = wmma_bf16(af, bf_, a1[i]);
        }
    }
    __syncthreads();
#pragma unroll
    for (int i = 0; i < 2; i++) {
        int n = (w * 2 + i) * 16 + lm;
#pragma unroll
        for (int r = 0; r < 8; r++) { int m = r + mh; xa[m * 128 + n] = f2bf(silu(a1[i][r] + b1[n])); }
    }
    __syncthreads();
    // GEMM2: 16x128 @ 128x384, 24 col tiles -> 6 per wave
    v8f a2[6];
#pragma unroll
    for (int i = 0; i < 6; i++) a2[i] = zero8();
    for (int kt = 0; kt < 4; kt++) {
        v16bf af = load_frag(xa + lm * 128 + kt * 32 + kh);
#pragma unroll
        for (int i = 0; i < 6; i++) {
            int n = (w * 6 + i) * 16 + lm;
            v16bf bf_ = load_frag(w2t + (size_t)n * 128 + kt * 32 + kh);
            a2[i] = wmma_bf16(af, bf_, a2[i]);
        }
    }
#pragma unroll
    for (int i = 0; i < 6; i++) {
        int n = (w * 6 + i) * 16 + lm;
#pragma unroll
        for (int r = 0; r < 8; r++) {
            int m = r + mh, g = r0 + m;
            if (g < N) xh[(size_t)g * 384 + n] = a2[i][r] + b2[n];
        }
    }
}

// ---------------- fused edge kernel (16 edges / block) ----------------
struct EdgeArgs {
    const float *xh, *edge_rbf, *weight, *edge_vector, *vec;
    const int* eidx;
    const __bf16 *dw1t, *dw2t, *rbfwt, *scalewt, *diagw1t;
    const float *dir_b1, *dir_b2, *rbf_b, *scale_b, *diag_b1, *diag_w2, *diag_b2;
    const float *kern_r, *kern_i, *UV, *fc_b;
    float *dxacc, *dvecacc;
    int E, N;
};

__global__ __launch_bounds__(128) void k_edge(EdgeArgs A) {
    __shared__ __align__(16) __bf16 aw[16 * 416];   // A staging: weight tile -> h -> rbfh(bf16) -> xe(bf16)
    __shared__ __align__(16) __bf16 rbfA[16 * 32];
    __shared__ __align__(16) float  buf[16 * 384];  // rbfh f32, later union: ri/t1/conv/a8/scal/ang
    __shared__ __align__(16) float  xhj[16 * 384];  // xh[src] * rbfh  (xe|xh2|xh3)
    __shared__ int sidx[32];                         // src[16], dst[16]

    float* ri  = buf;          // 16*192
    float* t1  = buf + 3072;   // 16*48
    float* cnv = buf + 3840;   // 16*16
    float* a8  = buf + 4096;   // 16*8
    float* scl = buf + 4224;   // 16*4
    float* ang = buf + 4288;   // 16*32

    int tid = threadIdx.x, e0 = blockIdx.x * 16;
    int lane = tid & 31, w = tid >> 5, lm = lane & 15, kh = (lane >> 4) * 8, mh = (lane >> 4) * 8;

    // ---- stage A1 (weight tile, f32->bf16), rbf tile, indices
    for (int i = tid; i < 16 * 416; i += 128) {
        int r = i / 416, c = i - r * 416, eg = e0 + r;
        aw[i] = (eg < A.E) ? f2bf(A.weight[(size_t)eg * 416 + c]) : f2bf(0.f);
    }
    for (int i = tid; i < 512; i += 128) {
        int r = i >> 5, c = i & 31, eg = e0 + r;
        rbfA[i] = (eg < A.E) ? f2bf(A.edge_rbf[(size_t)eg * 32 + c]) : f2bf(0.f);
    }
    if (tid < 16) {
        int eg = e0 + tid;
        sidx[tid]      = (eg < A.E) ? A.eidx[eg] : 0;
        sidx[16 + tid] = (eg < A.E) ? A.eidx[A.E + eg] : 0;
    }
    __syncthreads();

    // ---- GEMM1: 16x416 @ 416x384 (13 k-steps), 24 col tiles -> 6 per wave
    v8f acc[6];
#pragma unroll
    for (int i = 0; i < 6; i++) acc[i] = zero8();
    for (int kt = 0; kt < 13; kt++) {
        v16bf af = load_frag(aw + lm * 416 + kt * 32 + kh);
#pragma unroll
        for (int i = 0; i < 6; i++) {
            int n = (w * 6 + i) * 16 + lm;
            v16bf bf_ = load_frag(A.dw1t + (size_t)n * 416 + kt * 32 + kh);
            acc[i] = wmma_bf16(af, bf_, acc[i]);
        }
    }
    __syncthreads();
    // h = silu(. + dir_b1) -> aw (16x384 bf16)
#pragma unroll
    for (int i = 0; i < 6; i++) {
        int n = (w * 6 + i) * 16 + lm;
        float bb = A.dir_b1[n];
#pragma unroll
        for (int r = 0; r < 8; r++) { int m = r + mh; aw[m * 384 + n] = f2bf(silu(acc[i][r] + bb)); }
    }
    __syncthreads();

    // ---- rbf GEMM (K=32, one step) + GEMM2: 16x384 @ 384x384 (12 k-steps)
    v8f acr[6];
#pragma unroll
    for (int i = 0; i < 6; i++) acr[i] = zero8();
    {
        v16bf af = load_frag(rbfA + lm * 32 + kh);
#pragma unroll
        for (int i = 0; i < 6; i++) {
            int n = (w * 6 + i) * 16 + lm;
            v16bf bf_ = load_frag(A.rbfwt + (size_t)n * 32 + kh);
            acr[i] = wmma_bf16(af, bf_, acr[i]);
        }
    }
#pragma unroll
    for (int i = 0; i < 6; i++) acc[i] = zero8();
    for (int kt = 0; kt < 12; kt++) {
        v16bf af = load_frag(aw + lm * 384 + kt * 32 + kh);
#pragma unroll
        for (int i = 0; i < 6; i++) {
            int n = (w * 6 + i) * 16 + lm;
            v16bf bf_ = load_frag(A.dw2t + (size_t)n * 384 + kt * 32 + kh);
            acc[i] = wmma_bf16(af, bf_, acc[i]);
        }
    }
    __syncthreads();
    // rbfh = (rbfGEMM + rbf_b) * (GEMM2 + dir_b2) -> buf(f32) and aw(bf16)
#pragma unroll
    for (int i = 0; i < 6; i++) {
        int n = (w * 6 + i) * 16 + lm;
        float rb = A.rbf_b[n], d2 = A.dir_b2[n];
#pragma unroll
        for (int r = 0; r < 8; r++) {
            int m = r + mh;
            float rh = (acr[i][r] + rb) * (acc[i][r] + d2);
            buf[m * 384 + n] = rh;
            aw[m * 384 + n]  = f2bf(rh);
        }
    }
    __syncthreads();

    // ---- xh_j = xh[src] * rbfh (xe | xh2 | xh3)
    for (int i = tid; i < 6144; i += 128) {
        int m = i / 384, c = i - m * 384, s = sidx[m];
        xhj[i] = A.xh[(size_t)s * 384 + c] * buf[i];
    }
    __syncthreads();

    // ---- diag GEMM: 16x384 @ 384x48, 3 col tiles on waves 0..2
    if (w < 3) {
        v8f ad = zero8();
        for (int kt = 0; kt < 12; kt++) {
            v16bf af = load_frag(aw + lm * 384 + kt * 32 + kh);
            v16bf bf_ = load_frag(A.diagw1t + (size_t)(w * 16 + lm) * 384 + kt * 32 + kh);
            ad = wmma_bf16(af, bf_, ad);
        }
        int n = w * 16 + lm;
        float bb = A.diag_b1[n];
#pragma unroll
        for (int r = 0; r < 8; r++) { int m = r + mh; t1[m * 48 + n] = silu(ad[r] + bb); }
    }
    __syncthreads();

    // ---- xe (bf16) -> aw (16x128)
    for (int i = tid; i < 2048; i += 128) {
        int m = i >> 7, c = i & 127;
        aw[i] = f2bf(xhj[m * 384 + c]);
    }
    __syncthreads();

    // ---- scale GEMM: 16x128 @ 128x192, 12 col tiles -> 3 per wave
    v8f asx[3];
#pragma unroll
    for (int i = 0; i < 3; i++) asx[i] = zero8();
    for (int kt = 0; kt < 4; kt++) {
        v16bf af = load_frag(aw + lm * 128 + kt * 32 + kh);
#pragma unroll
        for (int i = 0; i < 3; i++) {
            int n = (w * 3 + i) * 16 + lm;
            v16bf bf_ = load_frag(A.scalewt + (size_t)n * 128 + kt * 32 + kh);
            asx[i] = wmma_bf16(af, bf_, asx[i]);
        }
    }
#pragma unroll
    for (int i = 0; i < 3; i++) {
        int n = (w * 3 + i) * 16 + lm;
        float sb = A.scale_b[n];
#pragma unroll
        for (int r = 0; r < 8; r++) { int m = r + mh; ri[m * 192 + n] = asx[i][r] + sb; }
    }
    __syncthreads();

    // ---- circular conv (17x6 taps) + diag_w2 layer; thread = (edge, k<8)
    {
        int e = tid >> 3, k = tid & 7;
        float sr = 0.f, si = 0.f, tr = 0.f, ti = 0.f;
        for (int l = 0; l < 6; l++) { // j==0: qr=1, qi=0
            sr += KSCALEf * A.kern_r[l * 8 + k];
            tr += KSCALEf * A.kern_i[l * 8 + k];
        }
        for (int j = 1; j < 17; j++)
            for (int l = 0; l < 6; l++) {
                float qr = ri[e * 192 + (j - 1) * 6 + l];
                float qi = ri[e * 192 + 96 + (j - 1) * 6 + l];
                float kr = KSCALEf * A.kern_r[j * 48 + l * 8 + k];
                float ki = KSCALEf * A.kern_i[j * 48 + l * 8 + k];
                sr += qr * kr; si += qi * ki; tr += qr * ki; ti += qi * kr;
            }
        cnv[e * 16 + k]     = sr - si;
        cnv[e * 16 + 8 + k] = tr + ti;
        float t2 = A.diag_b2[k];
        for (int c = 0; c < 48; c++) t2 += t1[e * 48 + c] * A.diag_w2[c * 8 + k];
        a8[e * 8 + k] = silu(t2);
    }
    __syncthreads();
    if (tid < 16) {
        int e = tid;
        float ca = 0.f, c1 = 0.f, ia = 0.f, i1 = 0.f;
        for (int k = 0; k < 8; k++) {
            float av = a8[e * 8 + k], cr = cnv[e * 16 + k], ci = cnv[e * 16 + 8 + k];
            ca += cr * av; c1 += cr; ia += ci * av; i1 += ci;
        }
        scl[e * 4] = ca; scl[e * 4 + 1] = c1; scl[e * 4 + 2] = ia; scl[e * 4 + 3] = i1;
    }
    __syncthreads();
    for (int i = tid; i < 512; i += 128) {
        int e = i >> 5, m = i & 31;
        float U = A.UV[m], V = A.UV[32 + m], fb = A.fc_b[m];
        float krr = scl[e * 4] * U + scl[e * 4 + 1] * V + fb;
        float kii = scl[e * 4 + 2] * U + scl[e * 4 + 3] * V + fb;
        ang[i] = atan2f(kii, krr);
    }
    __syncthreads();

    // ---- scatter: agg = [ang | xe] into dxacc
    for (int i = tid; i < 2560; i += 128) {
        int e = i / 160, c = i - e * 160, eg = e0 + e;
        if (eg >= A.E) continue;
        int d = sidx[16 + e];
        float v = (c < 32) ? ang[e * 32 + c] : xhj[e * 384 + (c - 32)];
        atomicAdd(A.dxacc + (size_t)d * 160 + c, v);
    }
    // ---- scatter: vmsg into dvecacc
    for (int i = tid; i < 6144; i += 128) {
        int e = i / 384, r = i - e * 384, eg = e0 + e;
        if (eg >= A.E) continue;
        int dd = r >> 7, c = r & 127, s = sidx[e], d = sidx[16 + e];
        float v = A.vec[(size_t)s * 384 + r] * (xhj[e * 384 + 128 + c] * INV_SQRT_3f)
                + xhj[e * 384 + 256 + c] * A.edge_vector[(size_t)eg * 3 + dd];
        atomicAdd(A.dvecacc + (size_t)d * 384 + r, v * INV_SQRT_Hf);
    }
}

// ---------------- final: LN(dx) -> cos/sin(dxs@s2), dy, dvec ----------------
__global__ __launch_bounds__(64) void k_final(
    const float* __restrict__ dxacc, const float* __restrict__ dvecacc,
    const float* __restrict__ g, const float* __restrict__ b,
    const float* __restrict__ s2w, const float* __restrict__ s2b,
    float* __restrict__ out, int N) {
    int n = blockIdx.x, t = threadIdx.x;
    __shared__ float row[160], rn[160], st[2];
    for (int i = t; i < 160; i += 64) row[i] = dxacc[(size_t)n * 160 + i];
    __syncthreads();
    if (t == 0) {
        float m = 0.f;
        for (int i = 0; i < 160; i++) m += row[i];
        m *= (1.f / 160.f);
        float v = 0.f;
        for (int i = 0; i < 160; i++) { float d = row[i] - m; v += d * d; }
        st[0] = m; st[1] = rsqrtf(v * (1.f / 160.f) + 1e-5f);
    }
    __syncthreads();
    for (int i = t; i < 160; i += 64) rn[i] = (row[i] - st[0]) * st[1] * g[i] + b[i];
    __syncthreads();
    float s = s2b[t];
    for (int k = 0; k < 32; k++) s += rn[k] * s2w[k * 64 + t];
    out[(size_t)n * 128 + 2 * t]     = cosf(s);
    out[(size_t)n * 128 + 2 * t + 1] = sinf(s);
    float* o2 = out + (size_t)N * 128;
    for (int i = t; i < 128; i += 64) o2[(size_t)n * 128 + i] = rn[32 + i];
    float* o3 = out + (size_t)N * 256;
    for (int i = t; i < 384; i += 64) o3[(size_t)n * 384 + i] = dvecacc[(size_t)n * 384 + i];
}

// ---------------- host launcher ----------------
extern "C" void kernel_launch(void* const* d_in, const int* in_sizes, int n_in,
                              void* d_out, int out_size, void* d_ws, size_t ws_size,
                              hipStream_t stream) {
    // inputs in setup_inputs() dict order (params nested dict in insertion order)
    const float* x           = (const float*)d_in[0];
    const float* vec         = (const float*)d_in[1];
    const float* edge_rbf    = (const float*)d_in[2];
    const float* weight      = (const float*)d_in[3];
    const float* edge_vector = (const float*)d_in[4];
    const int*   eidx        = (const int*)d_in[5];
    const float* ln_g   = (const float*)d_in[6];
    const float* ln_b   = (const float*)d_in[7];
    const float* xp_w1  = (const float*)d_in[8];
    const float* xp_b1  = (const float*)d_in[9];
    const float* xp_w2  = (const float*)d_in[10];
    const float* xp_b2  = (const float*)d_in[11];
    const float* rbf_w  = (const float*)d_in[12];
    const float* rbf_b  = (const float*)d_in[13];
    const float* dir_w1 = (const float*)d_in[14];
    const float* dir_b1 = (const float*)d_in[15];
    const float* dir_w2 = (const float*)d_in[16];
    const float* dir_b2 = (const float*)d_in[17];
    const float* scale_w = (const float*)d_in[18];
    const float* scale_b = (const float*)d_in[19];
    const float* diag_w1 = (const float*)d_in[20];
    const float* diag_b1 = (const float*)d_in[21];
    const float* diag_w2 = (const float*)d_in[22];
    const float* diag_b2 = (const float*)d_in[23];
    const float* diachi1 = (const float*)d_in[24];
    const float* kern_r  = (const float*)d_in[25];
    const float* kern_i  = (const float*)d_in[26];
    const float* dia_w   = (const float*)d_in[27];
    const float* dia_b   = (const float*)d_in[28];
    const float* fc_w    = (const float*)d_in[29];
    const float* fc_b    = (const float*)d_in[30];
    const float* dxln_g  = (const float*)d_in[31];
    const float* dxln_b  = (const float*)d_in[32];
    const float* s2_w    = (const float*)d_in[33];
    const float* s2_b    = (const float*)d_in[34];

    const int N = in_sizes[0] / 128;
    const int E = in_sizes[2] / 32;

    // workspace carve (d_ws is 256B aligned; all block sizes are multiples of 256B)
    char* base = (char*)d_ws;
    float* dxacc = (float*)base;   base += (size_t)N * 160 * 4;
    float* dvecacc = (float*)base; base += (size_t)N * 384 * 4;
    float* xh = (float*)base;      base += (size_t)N * 384 * 4;
    float* UV = (float*)base;      base += 256;
    __bf16* xpw1t  = (__bf16*)base; base += (size_t)128 * 128 * 2;
    __bf16* xpw2t  = (__bf16*)base; base += (size_t)384 * 128 * 2;
    __bf16* rbfwt  = (__bf16*)base; base += (size_t)384 * 32 * 2;
    __bf16* dirw1t = (__bf16*)base; base += (size_t)384 * 416 * 2;
    __bf16* dirw2t = (__bf16*)base; base += (size_t)384 * 384 * 2;
    __bf16* scwt   = (__bf16*)base; base += (size_t)192 * 128 * 2;
    __bf16* dgw1t  = (__bf16*)base; base += (size_t)48 * 384 * 2;
    (void)ws_size; (void)n_in; (void)out_size;

    // zero segment-sum accumulators (atomics accumulate into these each call)
    hipMemsetAsync(dxacc, 0, (size_t)N * (160 + 384) * 4, stream);

    // param prep
    auto cvt = [&](const float* W, __bf16* WT, int K, int Nc) {
        int tot = K * Nc;
        k_cvt_t<<<(tot + 255) / 256, 256, 0, stream>>>(W, WT, K, Nc);
    };
    cvt(xp_w1, xpw1t, 128, 128);
    cvt(xp_w2, xpw2t, 128, 384);
    cvt(rbf_w, rbfwt, 32, 384);
    cvt(dir_w1, dirw1t, 416, 384);
    cvt(dir_w2, dirw2t, 384, 384);
    cvt(scale_w, scwt, 128, 192);
    cvt(diag_w1, dgw1t, 384, 48);
    k_uv<<<1, 32, 0, stream>>>(diachi1, dia_w, dia_b, fc_w, UV);

    // node MLP
    k_node<<<(N + 15) / 16, 128, 0, stream>>>(x, ln_g, ln_b, xpw1t, xp_b1, xpw2t, xp_b2, xh, N);

    // fused edge pipeline
    EdgeArgs A;
    A.xh = xh; A.edge_rbf = edge_rbf; A.weight = weight; A.edge_vector = edge_vector; A.vec = vec;
    A.eidx = eidx;
    A.dw1t = dirw1t; A.dw2t = dirw2t; A.rbfwt = rbfwt; A.scalewt = scwt; A.diagw1t = dgw1t;
    A.dir_b1 = dir_b1; A.dir_b2 = dir_b2; A.rbf_b = rbf_b; A.scale_b = scale_b;
    A.diag_b1 = diag_b1; A.diag_w2 = diag_w2; A.diag_b2 = diag_b2;
    A.kern_r = kern_r; A.kern_i = kern_i; A.UV = UV; A.fc_b = fc_b;
    A.dxacc = dxacc; A.dvecacc = dvecacc; A.E = E; A.N = N;
    k_edge<<<(E + 15) / 16, 128, 0, stream>>>(A);

    // final node-wise epilogue
    k_final<<<N, 64, 0, stream>>>(dxacc, dvecacc, dxln_g, dxln_b, s2_w, s2_b, (float*)d_out, N);
}
